// AlphaStock_65068754534989
// MI455X (gfx1250) — matrix operand order repaired and checked by reading, not compiled, not run on hardware
//
#include <hip/hip_runtime.h>
#include <hip/hip_bf16.h>

typedef __attribute__((ext_vector_type(16))) _Float16 v16h;
typedef __attribute__((ext_vector_type(8)))  _Float16 v8h;
typedef __attribute__((ext_vector_type(8)))  float    v8f;

#define BB 2
#define MM 512
#define TT 64
#define FF 16
#define HH 128
#define NN (BB*MM)   // 1024
#define G4 (4*HH)    // 512

union V16H { v16h v; v8h h2[2]; _Float16 h[16]; };
union V8F  { v8f v; float f[8]; uint4 q[2]; };

// ---- WMMA fragment builders (gfx1250 wave32 layouts, CDNA5 ISA) ----

// A fragment: 16x32 f16 tile from row-major [16+][ldk] f16 buffer, K-offset ko.
// lane L: M = L%16; halves j: K = j + 16*(L/16)  -> two contiguous v8h loads
static __device__ __forceinline__ v16h load_a16(const _Float16* buf, int ldk, int ko) {
    int ln = threadIdx.x & 31;
    int m = ln & 15;
    int base = ko + ((ln >> 4) << 4);
    const _Float16* p = buf + m * ldk + base;
    V16H r;
    r.h2[0] = *(const v8h*)p;
    r.h2[1] = *(const v8h*)(p + 8);
    return r.v;
}

// B fragment: 32x16 f16 tile; weights stored row-major [n][ldk] (i.e. B^T storage = W[out][in]).
// lane L: N = n0 + L%16; halves j: K = ko + j + 16*(L/16)  -> one contiguous 32B load
static __device__ __forceinline__ v16h load_b16(const _Float16* W, int ldk, int n0, int ko) {
    int ln = threadIdx.x & 31;
    int n = n0 + (ln & 15);
    int base = ko + ((ln >> 4) << 4);
    return *(const v16h*)(W + n * ldk + base);
}

static __device__ __forceinline__ v8f wmma_f16(v16h a, v16h b, v8f c) {
    return __builtin_amdgcn_wmma_f32_16x16x32_f16(false, a, false, b, (short)0, c, false, false);
}

static __device__ __forceinline__ float sigm(float x) { return 1.f / (1.f + expf(-x)); }

// gate LDS layout: [tile ctg][lane][r] so a lane's v8f accumulator is one 32B store.
// value at (row 0..15, gate col cg 0..511):
//   ctg = cg>>4, lane = (cg&15) + 16*(row>>3), r = row&7
static __device__ __forceinline__ int gaddr(int row, int cg) {
    return ((cg >> 4) << 8) + (((cg & 15) + ((row >> 3) << 4)) << 3) + (row & 7);
}

// ---------------- conversion kernels ----------------

__global__ void k_f32_to_f16(const float* __restrict__ src, _Float16* __restrict__ dst, int n) {
    for (int i = blockIdx.x * blockDim.x + threadIdx.x; i < n; i += gridDim.x * blockDim.x)
        dst[i] = (_Float16)src[i];
}

// W_ih (512x16) -> f16 padded to K=32 with zeros; biasC = b_ih + b_hh
__global__ void k_prep_wih_bias(const float* __restrict__ Wih, const float* __restrict__ bih,
                                const float* __restrict__ bhh,
                                _Float16* __restrict__ wih16, float* __restrict__ biasC) {
    int i = blockIdx.x * blockDim.x + threadIdx.x;
    if (i < G4) biasC[i] = bih[i] + bhh[i];
    if (i < G4 * 32) {
        int g = i >> 5, f = i & 31;
        wih16[i] = (f < FF) ? (_Float16)Wih[g * FF + f] : (_Float16)0.f;
    }
}

// ---------------- fused LSTM recurrence ----------------
// grid = NN/16 workgroups, 512 threads (16 waves, wave32). Each wave owns 32 gate columns
// (2 col-tiles) and register-caches its 10 B fragments (80 VGPRs) for all 64 timesteps:
// per-step traffic is only the h A-fragments (ds_load_b128) and the gate writeback
// (ds_store_b128). ~150-180 VGPRs/wave -> fits without scratch spills.
__global__ void __launch_bounds__(512, 1)
k_lstm(const _Float16* __restrict__ x16,   // (N, T*F) f16
       const _Float16* __restrict__ wih16, // (512, 32) f16, K-padded
       const _Float16* __restrict__ whh16, // (512, 128) f16
       const float* __restrict__ biasC,    // (512) f32
       _Float16* __restrict__ rep)         // (N, T, H) f16
{
    extern __shared__ __align__(32) char smem[];
    _Float16* s_x = (_Float16*)smem;              // 16*32  f16 = 1KB
    _Float16* s_h = s_x + 16 * 32;                // 16*128 f16 = 4KB
    float*    s_c = (float*)(s_h + 16 * 128);     // 16*128 f32 = 8KB
    float*    s_g = s_c + 16 * 128;               // 16*512 f32 = 32KB (tile-major layout)

    const int tid = threadIdx.x;
    const int w   = tid >> 5;       // 0..15
    const int ln  = tid & 31;
    const int nbase = blockIdx.x * 16;
    const int col = ln & 15;

    for (int i = tid; i < 16 * 128; i += 512) { s_h[i] = (_Float16)0.f; s_c[i] = 0.f; }
    for (int i = tid; i < 16 * 16; i += 512) s_x[(i >> 4) * 32 + 16 + (i & 15)] = (_Float16)0.f;

    // register-cache this wave's B fragments and biases (weights are L2-resident, loaded once)
    v16h bih[2];
    v16h bhh[2][4];
    float bc[2];
#pragma unroll
    for (int ctl = 0; ctl < 2; ++ctl) {
        int n0 = (w * 2 + ctl) * 16;
        bc[ctl] = biasC[n0 + col];
        bih[ctl] = load_b16(wih16, 32, n0, 0);
#pragma unroll
        for (int ks = 0; ks < 4; ++ks)
            bhh[ctl][ks] = load_b16(whh16, 128, n0, ks * 32);
    }
    __syncthreads();

    for (int t = 0; t < TT; ++t) {
        // stage x tile (16 seqs x 16 feats) with 8B vector copies; pad half stays zero
        if (tid < 64) {
            int r = tid >> 2, seg = (tid & 3) << 2;
            *(uint2*)(s_x + r * 32 + seg) =
                *(const uint2*)(x16 + (size_t)(nbase + r) * (TT * FF) + t * FF + seg);
        }
        if (tid < 16 && t + 1 < TT)   // gfx1250 global_prefetch_b8 on next step's x rows
            __builtin_prefetch(x16 + (size_t)(nbase + tid) * (TT * FF) + (t + 1) * FF, 0, 3);

        // h A-fragments from previous step (stable until elementwise barrier)
        v16h ah0 = load_a16(s_h, 128, 0);
        v16h ah1 = load_a16(s_h, 128, 32);
        v16h ah2 = load_a16(s_h, 128, 64);
        v16h ah3 = load_a16(s_h, 128, 96);
        __syncthreads();
        v16h ax = load_a16(s_x, 32, 0);

        // 10 back-to-back WMMAs: all B operands already in VGPRs
#pragma unroll
        for (int ctl = 0; ctl < 2; ++ctl) {
            V8F acc;
#pragma unroll
            for (int r = 0; r < 8; ++r) acc.f[r] = bc[ctl];
            acc.v = wmma_f16(ax,  bih[ctl],    acc.v);
            acc.v = wmma_f16(ah0, bhh[ctl][0], acc.v);
            acc.v = wmma_f16(ah1, bhh[ctl][1], acc.v);
            acc.v = wmma_f16(ah2, bhh[ctl][2], acc.v);
            acc.v = wmma_f16(ah3, bhh[ctl][3], acc.v);
            // contiguous 32B gate writeback per lane
            float* gdst = s_g + (((w * 2 + ctl) << 8) + (ln << 3));
            *(uint4*)gdst       = acc.q[0];
            *(uint4*)(gdst + 4) = acc.q[1];
        }
        __syncthreads();

        // elementwise gate update (torch order i,f,g,o)
        for (int idx = tid; idx < 16 * 128; idx += 512) {
            int r = idx >> 7, hc = idx & 127;
            float gi = s_g[gaddr(r, hc)];
            float gf = s_g[gaddr(r, 128 + hc)];
            float gg = s_g[gaddr(r, 256 + hc)];
            float go = s_g[gaddr(r, 384 + hc)];
            float c = sigm(gf) * s_c[idx] + sigm(gi) * tanhf(gg);
            float h = sigm(go) * tanhf(c);
            s_c[idx] = c;
            s_h[idx] = (_Float16)h;
            rep[(size_t)(nbase + r) * (TT * HH) + t * HH + hc] = (_Float16)h;
        }
        __syncthreads();
    }
}

// ---------------- temporal attention (per sequence) ----------------
__global__ void k_temporal(const _Float16* __restrict__ rep, // (N,T,H) f16
                           const _Float16* __restrict__ w1f, // (H,H) f16
                           const float* __restrict__ b1,
                           const _Float16* __restrict__ w2f, // (H,H) f16
                           const float* __restrict__ b2,
                           const float* __restrict__ wa,     // (H)
                           _Float16* __restrict__ srep)      // (N,H) f16
{
    __shared__ float s_a2[128];
    __shared__ float s_alpha[64];
    const int tid = threadIdx.x;
    const int w = tid >> 5, ln = tid & 31;
    const int n = blockIdx.x;
    const _Float16* repn = rep + (size_t)n * TT * HH;

    // a2 = W2 * h_last + b2 (128-dim, one thread per output)
    {
        const _Float16* hl = repn + (TT - 1) * HH;
        const _Float16* wr = w2f + tid * HH;
        float acc = b2[tid];
        for (int h = 0; h < HH; ++h) acc += (float)hl[h] * (float)wr[h];
        s_a2[tid] = acc;
    }
    if (tid < 64) s_alpha[tid] = 0.f;
    __syncthreads();

    // a1 (WMMA) fused with alpha logits: wave w handles t-tile [16w,16w+16)
    {
        const int t0 = w * 16;
        const _Float16* arows = repn + (size_t)t0 * HH;
        v16h a0 = load_a16(arows, 128, 0);
        v16h a1 = load_a16(arows, 128, 32);
        v16h a2 = load_a16(arows, 128, 64);
        v16h a3 = load_a16(arows, 128, 96);
        const int col = ln & 15;
        const int rowoff = (ln >> 4) << 3;
        float rsum[8];
#pragma unroll
        for (int r = 0; r < 8; ++r) rsum[r] = 0.f;
        for (int ct = 0; ct < 8; ++ct) {
            int o0 = ct * 16;
            float bb = b1[o0 + col];
            V8F acc;
#pragma unroll
            for (int r = 0; r < 8; ++r) acc.f[r] = bb;
            acc.v = wmma_f16(a0, load_b16(w1f, 128, o0, 0),  acc.v);
            acc.v = wmma_f16(a1, load_b16(w1f, 128, o0, 32), acc.v);
            acc.v = wmma_f16(a2, load_b16(w1f, 128, o0, 64), acc.v);
            acc.v = wmma_f16(a3, load_b16(w1f, 128, o0, 96), acc.v);
            float wav = wa[o0 + col];
            float a2c = s_a2[o0 + col];
#pragma unroll
            for (int r = 0; r < 8; ++r) rsum[r] += wav * tanhf(acc.f[r] + a2c);
        }
#pragma unroll
        for (int r = 0; r < 8; ++r)
            atomicAdd(&s_alpha[t0 + r + rowoff], rsum[r]);   // ds_add_f32
    }
    __syncthreads();

    // softmax over T=64 (wave 0, shuffle reduction; +ba is softmax-invariant)
    if (w == 0) {
        float x0 = s_alpha[ln], x1 = s_alpha[ln + 32];
        float m = fmaxf(x0, x1);
        for (int off = 16; off; off >>= 1) m = fmaxf(m, __shfl_xor(m, off, 32));
        float e0 = expf(x0 - m), e1 = expf(x1 - m);
        float s = e0 + e1;
        for (int off = 16; off; off >>= 1) s += __shfl_xor(s, off, 32);
        float inv = 1.f / s;
        s_alpha[ln] = e0 * inv; s_alpha[ln + 32] = e1 * inv;
    }
    __syncthreads();

    // stock_rep = sum_t alpha[t] * rep[t,:]
    {
        float acc = 0.f;
        for (int t = 0; t < TT; ++t)
            acc += s_alpha[t] * (float)repn[t * HH + tid];
        srep[(size_t)n * HH + tid] = (_Float16)acc;
    }
}

// ---------------- q/k/v projections ----------------
// v is stored transposed (B,H,M) so the PV GEMM B-fragment is a contiguous 32B load.
__global__ void k_qkv(const _Float16* __restrict__ srep, // (N,H) f16
                      const _Float16* __restrict__ wqf, const float* __restrict__ bq,
                      const _Float16* __restrict__ wkf, const float* __restrict__ bk,
                      const _Float16* __restrict__ wvf, const float* __restrict__ bv,
                      _Float16* __restrict__ q, _Float16* __restrict__ k,
                      _Float16* __restrict__ vT)
{
    const int tid = threadIdx.x, w = tid >> 5, ln = tid & 31;
    const int r0 = blockIdx.x * 16;
    const _Float16* arows = srep + (size_t)r0 * HH;
    v16h a0 = load_a16(arows, 128, 0);
    v16h a1 = load_a16(arows, 128, 32);
    v16h a2 = load_a16(arows, 128, 64);
    v16h a3 = load_a16(arows, 128, 96);
    const int col = ln & 15, rowoff = (ln >> 4) << 3;

    for (int task = w; task < 24; task += 4) {
        int m = task >> 3, ct = task & 7, o0 = ct * 16;
        const _Float16* W = (m == 0) ? wqf : (m == 1) ? wkf : wvf;
        const float* bias = (m == 0) ? bq : (m == 1) ? bk : bv;
        float bb = bias[o0 + col];
        V8F acc;
#pragma unroll
        for (int r = 0; r < 8; ++r) acc.f[r] = bb;
        acc.v = wmma_f16(a0, load_b16(W, 128, o0, 0),  acc.v);
        acc.v = wmma_f16(a1, load_b16(W, 128, o0, 32), acc.v);
        acc.v = wmma_f16(a2, load_b16(W, 128, o0, 64), acc.v);
        acc.v = wmma_f16(a3, load_b16(W, 128, o0, 96), acc.v);
        if (m == 2) {
#pragma unroll
            for (int r = 0; r < 8; ++r) {
                int row = r0 + r + rowoff;
                int bidx = row >> 9, ai = row & 511;
                vT[((size_t)bidx * HH + o0 + col) * MM + ai] = (_Float16)acc.f[r];
            }
        } else {
            _Float16* dst = (m == 0) ? q : k;
#pragma unroll
            for (int r = 0; r < 8; ++r)
                dst[(size_t)(r0 + r + rowoff) * HH + o0 + col] = (_Float16)acc.f[r];
        }
    }
}

// ---------------- CAAN cross-asset attention ----------------
// grid = B * (M/16); 128 threads. s=QK^T/sqrt(H) (WMMA) -> softmax (LDS) -> P*V (WMMA) -> score.
__global__ void k_caan(const _Float16* __restrict__ q, const _Float16* __restrict__ k,
                       const _Float16* __restrict__ vT,
                       const float* __restrict__ ww, const float* __restrict__ bw,
                       float* __restrict__ out)
{
    extern __shared__ __align__(32) char smem[];
    float*    s_s    = (float*)smem;                  // 16*512 f32 = 32KB
    _Float16* s_p    = (_Float16*)(s_s + 16 * 512);   // 16*512 f16 = 16KB
    float*    s_attn = (float*)(s_p + 16 * 512);      // 16*128 f32 = 8KB
    float*    s_red  = s_attn + 16 * 128;             // 16*8
    float*    s_stat = s_red + 16 * 8;                // 16

    const int tid = threadIdx.x, w = tid >> 5, ln = tid & 31;
    const int bb = blockIdx.x >> 5, qt = blockIdx.x & 31;
    const int qrow0 = bb * MM + qt * 16;
    const int col = ln & 15, rowoff = (ln >> 4) << 3;
    const float scale = 0.088388347648318447f;  // 1/sqrt(128)

    // scores: wave w owns asset columns [128w, 128w+128)
    {
        const _Float16* arows = q + (size_t)qrow0 * HH;
        v16h a0 = load_a16(arows, 128, 0);
        v16h a1 = load_a16(arows, 128, 32);
        v16h a2 = load_a16(arows, 128, 64);
        v16h a3 = load_a16(arows, 128, 96);
        const _Float16* kb = k + (size_t)bb * MM * HH;
        for (int ct = 0; ct < 8; ++ct) {
            int n0 = w * 128 + ct * 16;
            V8F acc;
#pragma unroll
            for (int r = 0; r < 8; ++r) acc.f[r] = 0.f;
            acc.v = wmma_f16(a0, load_b16(kb, 128, n0, 0),  acc.v);
            acc.v = wmma_f16(a1, load_b16(kb, 128, n0, 32), acc.v);
            acc.v = wmma_f16(a2, load_b16(kb, 128, n0, 64), acc.v);
            acc.v = wmma_f16(a3, load_b16(kb, 128, n0, 96), acc.v);
#pragma unroll
            for (int r = 0; r < 8; ++r)
                s_s[(r + rowoff) * 512 + n0 + col] = acc.f[r] * scale;
        }
    }
    __syncthreads();

    // row softmax over 512 (16 rows x 8 chunks of 64)
    {
        const int row = tid & 15, ch = tid >> 4;
        const float* sr = s_s + row * 512 + ch * 64;
        float m = -1e30f;
        for (int i = 0; i < 64; ++i) m = fmaxf(m, sr[i]);
        s_red[row * 8 + ch] = m;
        __syncthreads();
        if (tid < 16) {
            float mm = s_red[tid * 8];
            for (int i = 1; i < 8; ++i) mm = fmaxf(mm, s_red[tid * 8 + i]);
            s_stat[tid] = mm;
        }
        __syncthreads();
        float rm = s_stat[row];
        _Float16* pr = s_p + row * 512 + ch * 64;
        float sum = 0.f;
        for (int i = 0; i < 64; ++i) { float e = expf(sr[i] - rm); pr[i] = (_Float16)e; sum += e; }
        s_red[row * 8 + ch] = sum;
        __syncthreads();
        if (tid < 16) {
            float ss = 0.f;
            for (int i = 0; i < 8; ++i) ss += s_red[tid * 8 + i];
            s_stat[tid] = 1.f / ss;
        }
        __syncthreads();
    }

    // attn = P * V : K=512 (16 steps), wave w owns h-columns [32w, 32w+32)
    {
        int n0a = w * 32, n0b = n0a + 16;
        const _Float16* vb = vT + (size_t)bb * HH * MM;
        V8F acc0, acc1;
#pragma unroll
        for (int r = 0; r < 8; ++r) { acc0.f[r] = 0.f; acc1.f[r] = 0.f; }
        for (int ks = 0; ks < 16; ++ks) {
            v16h a = load_a16(s_p, 512, ks * 32);
            acc0.v = wmma_f16(a, load_b16(vb, 512, n0a, ks * 32), acc0.v);
            acc1.v = wmma_f16(a, load_b16(vb, 512, n0b, ks * 32), acc1.v);
        }
#pragma unroll
        for (int r = 0; r < 8; ++r) {
            int row = r + rowoff;
            float inv = s_stat[row];
            s_attn[row * 128 + n0a + col] = acc0.f[r] * inv;
            s_attn[row * 128 + n0b + col] = acc1.f[r] * inv;
        }
    }
    __syncthreads();

    if (tid < 16) {
        float acc = bw[0];
        for (int h = 0; h < HH; ++h) acc += s_attn[tid * 128 + h] * ww[h];
        out[bb * MM + qt * 16 + tid] = acc;
    }
}

// ---------------- launch ----------------

extern "C" void kernel_launch(void* const* d_in, const int* in_sizes, int n_in,
                              void* d_out, int out_size, void* d_ws, size_t ws_size,
                              hipStream_t stream) {
    (void)in_sizes; (void)n_in; (void)out_size; (void)ws_size;
    const float* x    = (const float*)d_in[0];
    const float* W_ih = (const float*)d_in[1];
    const float* W_hh = (const float*)d_in[2];
    const float* b_ih = (const float*)d_in[3];
    const float* b_hh = (const float*)d_in[4];
    const float* w1   = (const float*)d_in[5];
    const float* b1   = (const float*)d_in[6];
    const float* w2   = (const float*)d_in[7];
    const float* b2   = (const float*)d_in[8];
    const float* wa   = (const float*)d_in[9];
    // d_in[10] = ba: uniform shift before softmax -> softmax-invariant, skipped
    const float* wq   = (const float*)d_in[11];
    const float* bq   = (const float*)d_in[12];
    const float* wk   = (const float*)d_in[13];
    const float* bk   = (const float*)d_in[14];
    const float* wv   = (const float*)d_in[15];
    const float* bv   = (const float*)d_in[16];
    const float* ww   = (const float*)d_in[17];
    const float* bw   = (const float*)d_in[18];

    char* ws = (char*)d_ws;
    constexpr size_t O_X    = 0;                       // (N,T*F) f16   2 MB
    constexpr size_t O_WIH  = 2097152;                 // (512,32) f16
    constexpr size_t O_WHH  = O_WIH + 32768;           // (512,128) f16
    constexpr size_t O_W1   = O_WHH + 131072;
    constexpr size_t O_W2   = O_W1 + 32768;
    constexpr size_t O_WQ   = O_W2 + 32768;
    constexpr size_t O_WK   = O_WQ + 32768;
    constexpr size_t O_WV   = O_WK + 32768;
    constexpr size_t O_BC   = O_WV + 32768;            // (512) f32
    constexpr size_t O_REP  = O_BC + 2048;             // (N,T,H) f16  16 MB
    constexpr size_t O_SREP = O_REP + 16777216;        // (N,H) f16
    constexpr size_t O_Q    = O_SREP + 262144;
    constexpr size_t O_K    = O_Q + 262144;
    constexpr size_t O_VT   = O_K + 262144;            // (B,H,M) f16

    _Float16* ws_x    = (_Float16*)(ws + O_X);
    _Float16* ws_wih  = (_Float16*)(ws + O_WIH);
    _Float16* ws_whh  = (_Float16*)(ws + O_WHH);
    _Float16* ws_w1   = (_Float16*)(ws + O_W1);
    _Float16* ws_w2   = (_Float16*)(ws + O_W2);
    _Float16* ws_wq   = (_Float16*)(ws + O_WQ);
    _Float16* ws_wk   = (_Float16*)(ws + O_WK);
    _Float16* ws_wv   = (_Float16*)(ws + O_WV);
    float*    ws_bc   = (float*)(ws + O_BC);
    _Float16* ws_rep  = (_Float16*)(ws + O_REP);
    _Float16* ws_srep = (_Float16*)(ws + O_SREP);
    _Float16* ws_q    = (_Float16*)(ws + O_Q);
    _Float16* ws_k    = (_Float16*)(ws + O_K);
    _Float16* ws_vT   = (_Float16*)(ws + O_VT);

    k_f32_to_f16<<<1024, 256, 0, stream>>>(x,    ws_x,   NN * TT * FF);
    k_f32_to_f16<<<256,  256, 0, stream>>>(W_hh, ws_whh, G4 * HH);
    k_f32_to_f16<<<64,   256, 0, stream>>>(w1,   ws_w1,  HH * HH);
    k_f32_to_f16<<<64,   256, 0, stream>>>(w2,   ws_w2,  HH * HH);
    k_f32_to_f16<<<64,   256, 0, stream>>>(wq,   ws_wq,  HH * HH);
    k_f32_to_f16<<<64,   256, 0, stream>>>(wk,   ws_wk,  HH * HH);
    k_f32_to_f16<<<64,   256, 0, stream>>>(wv,   ws_wv,  HH * HH);
    k_prep_wih_bias<<<64, 256, 0, stream>>>(W_ih, b_ih, b_hh, ws_wih, ws_bc);

    // fused LSTM: 45KB dynamic LDS (x 1K + h 4K + c 8K + gates 32K); weights in VGPRs
    k_lstm<<<NN / 16, 512, 46080, stream>>>(ws_x, ws_wih, ws_whh, ws_bc, ws_rep);

    k_temporal<<<NN, 128, 0, stream>>>(ws_rep, ws_w1, b1, ws_w2, b2, wa, ws_srep);

    k_qkv<<<NN / 16, 128, 0, stream>>>(ws_srep, ws_wq, bq, ws_wk, bk, ws_wv, bv,
                                       ws_q, ws_k, ws_vT);

    // 57.9KB dynamic LDS: s(32K) + p(16K) + attn(8K) + reductions
    k_caan<<<BB * (MM / 16), 128, 57920, stream>>>(ws_q, ws_k, ws_vT, ww, bw, (float*)d_out);
}
